// Attention_10239202033987
// MI455X (gfx1250) — matrix operand order repaired and checked by reading, not compiled, not run on hardware
//
#include <hip/hip_runtime.h>
#include <hip/hip_bf16.h>

typedef __bf16 bf16;
typedef __attribute__((ext_vector_type(16))) __bf16 v16bf;
typedef __attribute__((ext_vector_type(8)))  __bf16 v8bf;
typedef __attribute__((ext_vector_type(8)))  float  v8f;

#define N_HEADS 16
#define HEAD_DIM 128
#define Bsz 4
#define Tlen 2048
#define Cdim 2048
#define N3 6144

__device__ inline v8f zero8() {
    v8f z = {0.f,0.f,0.f,0.f,0.f,0.f,0.f,0.f};
    return z;
}

// WMMA bf16: D = A(16x32) * B(32x16) + C, f32 accumulate
__device__ inline v8f wmma_bf16(v16bf a, v16bf b, v8f c) {
    return __builtin_amdgcn_wmma_f32_16x16x32_bf16(
        false, a, false, b, (short)0, c, false, false);
}

// ---- gfx1250 async global->LDS copy, GVS mode: SGPR64 base + VGPR32 offset.
__device__ inline unsigned lds_off(const void* p) {
    return (unsigned)(uintptr_t)p;   // low 32 bits of generic LDS ptr = LDS byte offset
}
__device__ inline void async_ld16_s(unsigned ldsoff, unsigned voff, const void* sbase) {
    asm volatile("global_load_async_to_lds_b128 %0, %1, %2"
                 :: "v"(ldsoff), "v"(voff),
                    "s"((unsigned long long)(uintptr_t)sbase)
                 : "memory");
}
__device__ inline void wait_async0() {
    asm volatile("s_wait_asynccnt 0x0" ::: "memory");
}

// A-fragment (16x32 bf16 row-major, leading dim ld elements).
__device__ inline v16bf load_frag_a(const bf16* base, int ld) {
    const int lane = threadIdx.x & 31;
    const int m = lane & 15, kh = lane >> 4;
    const bf16* row = base + m * ld;
    union { v16bf v; v8bf h[2]; } u;
    u.h[0] = *(const v8bf*)(row + kh * 8);
    u.h[1] = *(const v8bf*)(row + 16 + kh * 8);
    return u.v;
}

// B-fragment (32x16 bf16) from a K-transposed tile stored [n][k] row-major.
__device__ inline v16bf load_frag_bt(const bf16* base, int ld) {
    const int lane = threadIdx.x & 31;
    const int n = lane & 15, kg = lane >> 4;
    const bf16* row = base + n * ld;
    union { v16bf v; v8bf h[2]; } u;
    u.h[0] = *(const v8bf*)(row + kg * 8);
    u.h[1] = *(const v8bf*)(row + 16 + kg * 8);
    return u.v;
}

// ---------------------------------------------------------------------------
// Prep kernel A: fp32 -> bf16 bulk convert (for x)
// ---------------------------------------------------------------------------
__global__ void __launch_bounds__(256) cvt_bf16_kernel(
    const float* __restrict__ src, bf16* __restrict__ dst)
{
    const size_t i = ((size_t)blockIdx.x * 256 + threadIdx.x) * 8;
    const float4 a = *(const float4*)(src + i);
    const float4 b = *(const float4*)(src + i + 4);
    v8bf o;
    o[0]=(bf16)a.x; o[1]=(bf16)a.y; o[2]=(bf16)a.z; o[3]=(bf16)a.w;
    o[4]=(bf16)b.x; o[5]=(bf16)b.y; o[6]=(bf16)b.z; o[7]=(bf16)b.w;
    *(v8bf*)(dst + i) = o;
}

// ---------------------------------------------------------------------------
// Prep kernel B: transpose + convert: dst[n][k] = (bf16)src[k][n]
// ---------------------------------------------------------------------------
__global__ void __launch_bounds__(256) transpose_cvt_kernel(
    const float* __restrict__ src, bf16* __restrict__ dst, int K, int N)
{
    __shared__ float tile[32][33];
    const int n0 = blockIdx.x * 32, k0 = blockIdx.y * 32;
    const int tx = threadIdx.x & 31, ty = threadIdx.x >> 5;
    #pragma unroll
    for (int i = 0; i < 4; ++i)
        tile[ty + i * 8][tx] = src[(size_t)(k0 + ty + i * 8) * N + n0 + tx];
    __syncthreads();
    #pragma unroll
    for (int i = 0; i < 4; ++i)
        dst[(size_t)(n0 + ty + i * 8) * K + k0 + tx] = (bf16)tile[tx][ty + i * 8];
}

// ---------------------------------------------------------------------------
// Kernel 1: qkv = xb @ WqkvT^T + bqkv (bf16), fused RoPE; scatter q/k/v.
// ---------------------------------------------------------------------------
__global__ void __launch_bounds__(256) qkv_rope_kernel(
    const bf16* __restrict__ xb, const bf16* __restrict__ WqkvT,
    const float* __restrict__ bqkv,
    bf16* __restrict__ qw, bf16* __restrict__ kw, bf16* __restrict__ vw)
{
    __shared__ bf16 As[2][128 * 32];   // [m][k]
    __shared__ bf16 Bs[2][128 * 32];   // [n][k]

    const int tid  = threadIdx.x;
    const int lane = tid & 31;
    const int w    = tid >> 5;
    const int row0 = blockIdx.y * 128;
    const int col0 = blockIdx.x * 128;
    const int m0   = (w & 3) * 32;
    const int n0   = (w >> 2) * 64;

    const unsigned asb = lds_off(&As[0][0]);
    const unsigned bsb = lds_off(&Bs[0][0]);

    unsigned ldsc[2], voff[2];
    #pragma unroll
    for (int p = 0; p < 2; ++p) {
        const int c = tid + p * 256;
        const int r = c >> 2, c8 = (c & 3) * 8;
        ldsc[p] = (unsigned)(r * 32 + c8) * 2u;
        voff[p] = (unsigned)(r * Cdim + c8) * 2u;
    }
    const bf16* aBase = xb + (size_t)row0 * Cdim;
    const bf16* bBase = WqkvT + (size_t)col0 * Cdim;

    v8f acc[2][4];
    #pragma unroll
    for (int i = 0; i < 2; ++i)
        #pragma unroll
        for (int j = 0; j < 4; ++j) acc[i][j] = zero8();

    #pragma unroll
    for (int p = 0; p < 2; ++p) {
        async_ld16_s(asb + ldsc[p], voff[p], aBase);
        async_ld16_s(bsb + ldsc[p], voff[p], bBase);
    }

    #pragma clang loop unroll(disable)
    for (int kt = 0; kt < Cdim / 32; ++kt) {
        const unsigned cbuf = (unsigned)(kt & 1) * 8192u;
        wait_async0();
        __syncthreads();
        if (kt + 1 < Cdim / 32) {
            const unsigned nbuf = cbuf ^ 8192u;
            const bf16* sA = aBase + (kt + 1) * 32;
            const bf16* sB = bBase + (kt + 1) * 32;
            #pragma unroll
            for (int p = 0; p < 2; ++p) {
                async_ld16_s(asb + nbuf + ldsc[p], voff[p], sA);
                async_ld16_s(bsb + nbuf + ldsc[p], voff[p], sB);
            }
        }

        const bf16* a_s = &As[0][0] + (cbuf >> 1);   // bytes -> elements
        const bf16* b_s = &Bs[0][0] + (cbuf >> 1);
        // hoist all fragments, then 8 back-to-back WMMAs (no interleaved loads)
        v16bf a0 = load_frag_a(a_s + m0 * 32, 32);
        v16bf a1 = load_frag_a(a_s + (m0 + 16) * 32, 32);
        v16bf b0 = load_frag_bt(b_s + (n0 +  0) * 32, 32);
        v16bf b1 = load_frag_bt(b_s + (n0 + 16) * 32, 32);
        v16bf b2 = load_frag_bt(b_s + (n0 + 32) * 32, 32);
        v16bf b3 = load_frag_bt(b_s + (n0 + 48) * 32, 32);
        acc[0][0] = wmma_bf16(a0, b0, acc[0][0]);
        acc[1][0] = wmma_bf16(a1, b0, acc[1][0]);
        acc[0][1] = wmma_bf16(a0, b1, acc[0][1]);
        acc[1][1] = wmma_bf16(a1, b1, acc[1][1]);
        acc[0][2] = wmma_bf16(a0, b2, acc[0][2]);
        acc[1][2] = wmma_bf16(a1, b2, acc[1][2]);
        acc[0][3] = wmma_bf16(a0, b3, acc[0][3]);
        acc[1][3] = wmma_bf16(a1, b3, acc[1][3]);
    }

    // Epilogue: bias + RoPE + scatter (bf16)
    const float LN_BASE = 9.210340371976184f;   // ln(10000)
    #pragma unroll
    for (int i = 0; i < 2; ++i) {
        #pragma unroll
        for (int j = 0; j < 4; ++j) {
            const int col  = col0 + n0 + j * 16 + (lane & 15);
            const float bias = bqkv[col];
            #pragma unroll
            for (int r = 0; r < 8; ++r) {
                const int row = row0 + m0 + i * 16 + r + ((lane >> 4) << 3);
                float v = acc[i][j][r] + bias;
                float partner = __shfl_xor(v, 1, 32);   // col ^ 1 neighbor
                const int t     = row & (Tlen - 1);
                const int d128  = col & (HEAD_DIM - 1);
                const int pair  = d128 >> 1;
                const float inv_freq = __expf(-LN_BASE * (float)(2 * pair) * (1.0f / 128.0f));
                const float ang = (float)t * inv_freq;
                const float sn = __sinf(ang), cs = __cosf(ang);
                float roped;
                if ((col & 1) == 0) roped = v * cs - partner * sn;
                else                roped = partner * sn + v * cs;
                const int which = col >> 11;            // 0:q 1:k 2:v
                const float outv = (which == 2) ? v : roped;
                const int b_ = row >> 11;
                const int h  = (col & (Cdim - 1)) >> 7;
                const bf16 o = (bf16)outv;
                if (which == 0) {
                    qw[(((size_t)b_ * N_HEADS + h) * Tlen + t) * HEAD_DIM + d128] = o;
                } else if (which == 1) {
                    kw[(((size_t)b_ * N_HEADS + h) * Tlen + t) * HEAD_DIM + d128] = o;
                } else {
                    vw[(((size_t)b_ * N_HEADS + h) * HEAD_DIM + d128) * Tlen + t] = o;
                }
            }
        }
    }
}

// ---------------------------------------------------------------------------
// Kernel 2: causal flash attention.
// ---------------------------------------------------------------------------
__global__ void __launch_bounds__(256) attn_kernel(
    const bf16* __restrict__ qw, const bf16* __restrict__ kw,
    const bf16* __restrict__ vw, bf16* __restrict__ attn)
{
    __shared__ bf16 Ks[2][128 * 128];     // [s][d]
    __shared__ bf16 Vt[2][128 * 128];     // [d][s]
    __shared__ bf16 Ps[8][16 * 128];      // per-wave P staging [m][s]

    const int tid  = threadIdx.x;
    const int lane = tid & 31;
    const int w    = tid >> 5;
    const int qt   = blockIdx.x;
    const int bh   = blockIdx.y;
    const int b_   = bh >> 4, h = bh & 15;

    const bf16* Q   = qw + (size_t)bh * Tlen * HEAD_DIM;
    const bf16* K   = kw + (size_t)bh * Tlen * HEAD_DIM;
    const bf16* Vtg = vw + (size_t)bh * HEAD_DIM * Tlen;   // [d][t]
    const int q0 = qt * 128 + w * 16;

    const unsigned ksb = lds_off(&Ks[0][0]);
    const unsigned vtb = lds_off(&Vt[0][0]);

    unsigned ldsc[8], voffV[8];
    #pragma unroll
    for (int p = 0; p < 8; ++p) {
        const int c = tid + p * 256;
        const int r = c >> 4;            // row 0..127
        const int c8 = (c & 15) * 8;     // elem 0..120
        ldsc[p]  = (unsigned)(r * 128 + c8) * 2u;   // also the K global offset
        voffV[p] = (unsigned)(r * Tlen + c8) * 2u;
    }

    v16bf qf[4];
    #pragma unroll
    for (int c = 0; c < 4; ++c)
        qf[c] = load_frag_a(Q + (size_t)q0 * HEAD_DIM + c * 32, HEAD_DIM);

    v8f o[8];
    #pragma unroll
    for (int dt = 0; dt < 8; ++dt) o[dt] = zero8();
    float mrow[8], lrow[8];
    #pragma unroll
    for (int r = 0; r < 8; ++r) { mrow[r] = -3.0e38f; lrow[r] = 0.f; }

    const float scale = 0.08838834764831845f;   // 1/sqrt(128)

    #pragma unroll
    for (int p = 0; p < 8; ++p) {
        async_ld16_s(ksb + ldsc[p], ldsc[p], K);
        async_ld16_s(vtb + ldsc[p], voffV[p], Vtg);
    }

    #pragma clang loop unroll(disable)
    for (int kb = 0; kb <= qt; ++kb) {
        const unsigned cbuf = (unsigned)(kb & 1) * 32768u;
        wait_async0();
        __syncthreads();
        if (kb + 1 <= qt) {
            const unsigned nbuf = cbuf ^ 32768u;
            const bf16* sK = K + (size_t)(kb + 1) * 128 * HEAD_DIM;
            const bf16* sV = Vtg + (size_t)(kb + 1) * 128;
            #pragma unroll
            for (int p = 0; p < 8; ++p) {
                async_ld16_s(ksb + nbuf + ldsc[p], ldsc[p], sK);
                async_ld16_s(vtb + nbuf + ldsc[p], voffV[p], sV);
            }
        }

        const bf16* k_s = &Ks[0][0] + (cbuf >> 1);
        const bf16* v_s = &Vt[0][0] + (cbuf >> 1);
        const bool diag = (kb == qt);

        // S = Q K^T : 8 tiles of 16x16, contraction over d (4 x 32)
        v8f s_acc[8];
        #pragma unroll
        for (int nt = 0; nt < 8; ++nt) {
            v8f a = zero8();
            const bool skip = diag && (nt > w);
            if (!skip) {
                v16bf kb0 = load_frag_bt(k_s + (nt * 16) * 128 +  0, 128);
                v16bf kb1 = load_frag_bt(k_s + (nt * 16) * 128 + 32, 128);
                v16bf kb2 = load_frag_bt(k_s + (nt * 16) * 128 + 64, 128);
                v16bf kb3 = load_frag_bt(k_s + (nt * 16) * 128 + 96, 128);
                a = wmma_bf16(qf[0], kb0, a);
                a = wmma_bf16(qf[1], kb1, a);
                a = wmma_bf16(qf[2], kb2, a);
                a = wmma_bf16(qf[3], kb3, a);
            }
            s_acc[nt] = a;
        }

        // scale + causal mask + per-row max
        float rmax[8];
        #pragma unroll
        for (int r = 0; r < 8; ++r) rmax[r] = -3.0e38f;
        #pragma unroll
        for (int nt = 0; nt < 8; ++nt) {
            const bool skip = diag && (nt > w);
            #pragma unroll
            for (int r = 0; r < 8; ++r) {
                float v = s_acc[nt][r] * scale;
                if (skip) {
                    v = -3.0e38f;
                } else if (diag) {
                    const int sg = kb * 128 + nt * 16 + (lane & 15);
                    const int qg = qt * 128 + w * 16 + r + ((lane >> 4) << 3);
                    if (sg > qg) v = -3.0e38f;
                }
                s_acc[nt][r] = v;
                rmax[r] = fmaxf(rmax[r], v);
            }
        }
        #pragma unroll
        for (int r = 0; r < 8; ++r) {
            float v = rmax[r];
            v = fmaxf(v, __shfl_xor(v, 1, 32));
            v = fmaxf(v, __shfl_xor(v, 2, 32));
            v = fmaxf(v, __shfl_xor(v, 4, 32));
            v = fmaxf(v, __shfl_xor(v, 8, 32));
            rmax[r] = v;
        }

        float rsum[8];
        #pragma unroll
        for (int r = 0; r < 8; ++r) {
            const float newm = fmaxf(mrow[r], rmax[r]);
            const float sc   = __expf(mrow[r] - newm);
            mrow[r] = newm;
            lrow[r] *= sc;
            rsum[r] = 0.f;
            #pragma unroll
            for (int dt = 0; dt < 8; ++dt) o[dt][r] *= sc;
        }

        // P = exp(S - m) -> wave-private LDS for C->A layout conversion
        bf16* pw = &Ps[w][0];
        #pragma unroll
        for (int nt = 0; nt < 8; ++nt) {
            #pragma unroll
            for (int r = 0; r < 8; ++r) {
                const float p = __expf(s_acc[nt][r] - mrow[r]);
                rsum[r] += p;
                pw[(r + ((lane >> 4) << 3)) * 128 + nt * 16 + (lane & 15)] = (bf16)p;
            }
        }
        #pragma unroll
        for (int r = 0; r < 8; ++r) {
            float v = rsum[r];
            v += __shfl_xor(v, 1, 32);
            v += __shfl_xor(v, 2, 32);
            v += __shfl_xor(v, 4, 32);
            v += __shfl_xor(v, 8, 32);
            lrow[r] += v;
        }

        // O += P V : contraction over s (4 x 32), 8 output d-tiles
        v16bf pf[4];
        #pragma unroll
        for (int c = 0; c < 4; ++c)
            pf[c] = load_frag_a(pw + c * 32, 128);
        #pragma unroll
        for (int dt = 0; dt < 8; ++dt) {
            v16bf vb0 = load_frag_bt(v_s + (dt * 16) * 128 +  0, 128);
            v16bf vb1 = load_frag_bt(v_s + (dt * 16) * 128 + 32, 128);
            v16bf vb2 = load_frag_bt(v_s + (dt * 16) * 128 + 64, 128);
            v16bf vb3 = load_frag_bt(v_s + (dt * 16) * 128 + 96, 128);
            o[dt] = wmma_bf16(pf[0], vb0, o[dt]);
            o[dt] = wmma_bf16(pf[1], vb1, o[dt]);
            o[dt] = wmma_bf16(pf[2], vb2, o[dt]);
            o[dt] = wmma_bf16(pf[3], vb3, o[dt]);
        }
    }

    // finalize: O /= l, write bf16 row-major [B*T][C]
    #pragma unroll
    for (int r = 0; r < 8; ++r) {
        const float inv = 1.0f / lrow[r];
        const size_t row = (size_t)b_ * Tlen + qt * 128 + w * 16 + r + ((lane >> 4) << 3);
        #pragma unroll
        for (int dt = 0; dt < 8; ++dt) {
            const float v = o[dt][r] * inv;
            attn[row * Cdim + h * HEAD_DIM + dt * 16 + (lane & 15)] = (bf16)v;
        }
    }
}

// ---------------------------------------------------------------------------
// Kernel 3: out = attn @ WoutT^T + bout  (bf16 x bf16 -> fp32)
// ---------------------------------------------------------------------------
__global__ void __launch_bounds__(256) out_proj_kernel(
    const bf16* __restrict__ attn, const bf16* __restrict__ WoutT,
    const float* __restrict__ bout, float* __restrict__ out)
{
    __shared__ bf16 As[2][128 * 32];
    __shared__ bf16 Bs[2][128 * 32];

    const int tid  = threadIdx.x;
    const int lane = tid & 31;
    const int w    = tid >> 5;
    const int row0 = blockIdx.y * 128;
    const int col0 = blockIdx.x * 128;
    const int m0   = (w & 3) * 32;
    const int n0   = (w >> 2) * 64;

    const unsigned asb = lds_off(&As[0][0]);
    const unsigned bsb = lds_off(&Bs[0][0]);

    unsigned ldsc[2], voff[2];
    #pragma unroll
    for (int p = 0; p < 2; ++p) {
        const int c = tid + p * 256;
        const int r = c >> 2, c8 = (c & 3) * 8;
        ldsc[p] = (unsigned)(r * 32 + c8) * 2u;
        voff[p] = (unsigned)(r * Cdim + c8) * 2u;
    }
    const bf16* aBase = attn + (size_t)row0 * Cdim;
    const bf16* bBase = WoutT + (size_t)col0 * Cdim;

    v8f acc[2][4];
    #pragma unroll
    for (int i = 0; i < 2; ++i)
        #pragma unroll
        for (int j = 0; j < 4; ++j) acc[i][j] = zero8();

    #pragma unroll
    for (int p = 0; p < 2; ++p) {
        async_ld16_s(asb + ldsc[p], voff[p], aBase);
        async_ld16_s(bsb + ldsc[p], voff[p], bBase);
    }

    #pragma clang loop unroll(disable)
    for (int kt = 0; kt < Cdim / 32; ++kt) {
        const unsigned cbuf = (unsigned)(kt & 1) * 8192u;
        wait_async0();
        __syncthreads();
        if (kt + 1 < Cdim / 32) {
            const unsigned nbuf = cbuf ^ 8192u;
            const bf16* sA = aBase + (kt + 1) * 32;
            const bf16* sB = bBase + (kt + 1) * 32;
            #pragma unroll
            for (int p = 0; p < 2; ++p) {
                async_ld16_s(asb + nbuf + ldsc[p], voff[p], sA);
                async_ld16_s(bsb + nbuf + ldsc[p], voff[p], sB);
            }
        }

        const bf16* a_s = &As[0][0] + (cbuf >> 1);
        const bf16* b_s = &Bs[0][0] + (cbuf >> 1);
        v16bf a0 = load_frag_a(a_s + m0 * 32, 32);
        v16bf a1 = load_frag_a(a_s + (m0 + 16) * 32, 32);
        v16bf b0 = load_frag_bt(b_s + (n0 +  0) * 32, 32);
        v16bf b1 = load_frag_bt(b_s + (n0 + 16) * 32, 32);
        v16bf b2 = load_frag_bt(b_s + (n0 + 32) * 32, 32);
        v16bf b3 = load_frag_bt(b_s + (n0 + 48) * 32, 32);
        acc[0][0] = wmma_bf16(a0, b0, acc[0][0]);
        acc[1][0] = wmma_bf16(a1, b0, acc[1][0]);
        acc[0][1] = wmma_bf16(a0, b1, acc[0][1]);
        acc[1][1] = wmma_bf16(a1, b1, acc[1][1]);
        acc[0][2] = wmma_bf16(a0, b2, acc[0][2]);
        acc[1][2] = wmma_bf16(a1, b2, acc[1][2]);
        acc[0][3] = wmma_bf16(a0, b3, acc[0][3]);
        acc[1][3] = wmma_bf16(a1, b3, acc[1][3]);
    }

    #pragma unroll
    for (int i = 0; i < 2; ++i) {
        #pragma unroll
        for (int j = 0; j < 4; ++j) {
            const int col = col0 + n0 + j * 16 + (lane & 15);
            const float bias = bout[col];
            #pragma unroll
            for (int r = 0; r < 8; ++r) {
                const int row = row0 + m0 + i * 16 + r + ((lane >> 4) << 3);
                out[(size_t)row * Cdim + col] = acc[i][j][r] + bias;
            }
        }
    }
}

// ---------------------------------------------------------------------------
extern "C" void kernel_launch(void* const* d_in, const int* in_sizes, int n_in,
                              void* d_out, int out_size, void* d_ws, size_t ws_size,
                              hipStream_t stream) {
    const float* x    = (const float*)d_in[0];
    const float* Wqkv = (const float*)d_in[1];
    const float* bqkv = (const float*)d_in[2];
    const float* Wout = (const float*)d_in[3];
    const float* bout = (const float*)d_in[4];
    float* out = (float*)d_out;

    const size_t per = (size_t)Bsz * N_HEADS * Tlen * HEAD_DIM;  // 16.78M elems
    bf16* qw    = (bf16*)d_ws;
    bf16* kw    = qw + per;
    bf16* vw    = kw + per;          // [b][h][d][t] (transposed)
    bf16* attn  = vw + per;
    bf16* xb    = attn + per;                        // B*T*C
    bf16* WqkvT = xb + (size_t)Bsz * Tlen * Cdim;    // [N3][C]
    bf16* WoutT = WqkvT + (size_t)N3 * Cdim;         // [C][C]

    cvt_bf16_kernel<<<(unsigned)(((size_t)Bsz * Tlen * Cdim) / (256 * 8)), 256, 0, stream>>>(x, xb);
    transpose_cvt_kernel<<<dim3(N3 / 32, Cdim / 32), 256, 0, stream>>>(Wqkv, WqkvT, Cdim, N3);
    transpose_cvt_kernel<<<dim3(Cdim / 32, Cdim / 32), 256, 0, stream>>>(Wout, WoutT, Cdim, Cdim);

    qkv_rope_kernel<<<dim3(N3 / 128, (Bsz * Tlen) / 128), 256, 0, stream>>>(
        xb, WqkvT, bqkv, qw, kw, vw);
    attn_kernel<<<dim3(Tlen / 128, Bsz * N_HEADS), 256, 0, stream>>>(
        qw, kw, vw, attn);
    out_proj_kernel<<<dim3(Cdim / 128, (Bsz * Tlen) / 128), 256, 0, stream>>>(
        attn, WoutT, bout, out);
}
